// Unified_attention_46110768890193
// MI455X (gfx1250) — matrix operand order repaired and checked by reading, not compiled, not run on hardware
//
#include <hip/hip_runtime.h>
#include <hip/hip_bf16.h>

// Problem constants (match reference)
#define NN 8192
#define DD 256
#define EPSV 1e-6f
#define BPAD 40   // bf16 elements per LDS tile row (32 data + 8 pad -> 80B rows, no bank conflicts)

typedef __bf16 bf16_t;
typedef __attribute__((ext_vector_type(16))) __bf16 v16bf;
typedef __attribute__((ext_vector_type(8)))  __bf16 v8bf;
typedef __attribute__((ext_vector_type(4)))  __bf16 v4bf;
typedef __attribute__((ext_vector_type(8)))  float  v8f;
typedef __attribute__((ext_vector_type(4)))  int    i4;

__device__ __forceinline__ v16bf cat8(v8bf lo, v8bf hi) {
  return __builtin_shufflevector(lo, hi, 0,1,2,3,4,5,6,7,8,9,10,11,12,13,14,15);
}

// CDNA5 async global->LDS copy path (tracked by ASYNCcnt), with portable fallback.
// Builtin signature (from clang diagnostic): (global int4*, local int4*, imm offset, imm cpol)
#if __has_builtin(__builtin_amdgcn_global_load_async_to_lds_b128) && \
    __has_builtin(__builtin_amdgcn_s_wait_asynccnt)
#define USE_ASYNC_LDS 1
#define ASYNC_COPY16(g, d)                                                \
  __builtin_amdgcn_global_load_async_to_lds_b128(                         \
      (__attribute__((address_space(1))) i4*)(g),                         \
      (__attribute__((address_space(3))) i4*)(d), 0, 0)
#define WAIT_ASYNC(n) __builtin_amdgcn_s_wait_asynccnt(n)
#else
#define USE_ASYNC_LDS 0
#define ASYNC_COPY16(g, d) do { v8bf t_ = *(const v8bf*)(g); *(v8bf*)(d) = t_; } while (0)
#define WAIT_ASYNC(n) do { } while (0)
#endif

// ---------------------------------------------------------------------------
// Convert z (f32, row-major [N,D]) -> bf16
__global__ void k_cvt_z(const float* __restrict__ z, bf16_t* __restrict__ zb) {
  unsigned i = blockIdx.x * blockDim.x + threadIdx.x;   // over N*D/4
  float4 v = ((const float4*)z)[i];
  v4bf o = { (__bf16)v.x, (__bf16)v.y, (__bf16)v.z, (__bf16)v.w };
  ((v4bf*)zb)[i] = o;
}

// Transpose + convert W (f32 [D,D] row-major) -> Wt bf16 with Wt[n*D+k] = W[k*D+n]
__global__ void k_cvt_wt(const float* __restrict__ W, bf16_t* __restrict__ Wt) {
  unsigned idx = blockIdx.x * blockDim.x + threadIdx.x; // over D*D
  unsigned n = idx >> 8;
  unsigned k = idx & 255;
  Wt[idx] = (__bf16)W[k * DD + n];
}

// Row sums of Q -> qrsum[N]. One block per row, float4 streaming.
__global__ __launch_bounds__(256) void k_qrowsum(const float* __restrict__ Q,
                                                 float* __restrict__ qrsum) {
  __shared__ float red[256];
  const unsigned row = blockIdx.x;
  const float4* qr = (const float4*)Q + (size_t)row * (NN / 4);
  float s = 0.f;
  #pragma unroll
  for (int j = 0; j < (NN / 4) / 256; ++j) {
    float4 v = qr[threadIdx.x + j * 256];
    s += v.x + v.y + v.z + v.w;
  }
  red[threadIdx.x] = s;
  __syncthreads();
  for (int off = 128; off > 0; off >>= 1) {
    if ((int)threadIdx.x < off) red[threadIdx.x] += red[threadIdx.x + off];
    __syncthreads();
  }
  if (threadIdx.x == 0) qrsum[row] = red[0];
}

// ---------------------------------------------------------------------------
// Small GEMM: Zw_bf16 = bf16( z_bf16 @ W ).  Direct loads (operands tiny, L2).
__global__ __launch_bounds__(256) void k_gemm_zw(const bf16_t* __restrict__ A,
                                                 const bf16_t* __restrict__ Bt,
                                                 bf16_t* __restrict__ Zw) {
  const unsigned lane = threadIdx.x & 31;
  const unsigned wave = threadIdx.x >> 5;
  const unsigned half = lane >> 4;
  const unsigned lm   = lane & 15;
  const unsigned rowBase = blockIdx.y * 128 + wave * 16;
  const unsigned colBase = blockIdx.x * 128;

  const v8f zero = {0.f,0.f,0.f,0.f,0.f,0.f,0.f,0.f};
  v8f acc[8];
  #pragma unroll
  for (int ct = 0; ct < 8; ++ct) acc[ct] = zero;

  #pragma unroll
  for (unsigned kb = 0; kb < DD; kb += 32) {
    const unsigned abase = (rowBase + lm) * DD + kb + half * 16;
    v8bf a0 = *(const v8bf*)(A + abase);
    v8bf a1 = *(const v8bf*)(A + abase + 8);
    v16bf af = cat8(a0, a1);
    #pragma unroll
    for (unsigned ct = 0; ct < 8; ++ct) {
      const unsigned bbase = (colBase + ct * 16 + lm) * DD + kb + half * 16;
      v8bf b0 = *(const v8bf*)(Bt + bbase);
      v8bf b1 = *(const v8bf*)(Bt + bbase + 8);
      acc[ct] = __builtin_amdgcn_wmma_f32_16x16x32_bf16(
          false, af, false, cat8(b0, b1), (short)0, acc[ct], false, false);
    }
  }
  #pragma unroll
  for (unsigned ct = 0; ct < 8; ++ct) {
    const unsigned n = colBase + ct * 16 + lm;
    #pragma unroll
    for (unsigned r = 0; r < 8; ++r) {
      const unsigned m = rowBase + 8 * half + r;
      Zw[m * DD + n] = (__bf16)acc[ct][r];
    }
  }
}

// ---------------------------------------------------------------------------
// Main fused kernel: sim tile = Zw @ z^T via bf16 WMMA (f32 accum).
// B tile (128 cols x 32 k, bf16) staged in LDS, double-buffered with CDNA5
// async global->LDS copies (ASYNCcnt): fills for k-step i+1 overlap the WMMA
// burst on k-step i, and B is loaded from L2 once per block instead of once
// per wave (8x traffic cut). Epilogue fused; Q stream prefetched + NT loads.
__global__ __launch_bounds__(256) void k_main(const bf16_t* __restrict__ Zw,
                                              const bf16_t* __restrict__ Zb,
                                              const float* __restrict__ beta,
                                              const float* __restrict__ Q,
                                              const float* __restrict__ qrsum,
                                              float* __restrict__ out) {
  __shared__ __align__(32) bf16_t sB[2][128 * BPAD];

  const unsigned lane = threadIdx.x & 31;
  const unsigned wave = threadIdx.x >> 5;
  const unsigned half = lane >> 4;
  const unsigned lm   = lane & 15;
  const unsigned rowBase = blockIdx.y * 128 + wave * 16;
  const unsigned colBase = blockIdx.x * 128;

  // Prefetch this block's 128x128 Q tile (512 x 128B lines) into cache.
  #pragma unroll
  for (int p = 0; p < 2; ++p) {
    const unsigned l  = threadIdx.x + p * 256;          // line index 0..511
    const unsigned rr = blockIdx.y * 128 + (l >> 2);
    const unsigned cc = blockIdx.x * 128 + (l & 3) * 32;
    __builtin_prefetch(Q + (size_t)rr * NN + cc, 0, 0);
  }

  // Cooperative fill assignment: thread t moves 32B = 16 bf16 of the B tile:
  // row fc = t>>1 (column of sim), k-chunk fh = (t&1)*16.
  const unsigned fc = threadIdx.x >> 1;
  const unsigned fh = (threadIdx.x & 1) * 16;
  const bf16_t* gfill = Zb + (colBase + fc) * DD + fh;
  bf16_t* sfill0 = &sB[0][fc * BPAD + fh];
  bf16_t* sfill1 = &sB[1][fc * BPAD + fh];

  const v8f zero = {0.f,0.f,0.f,0.f,0.f,0.f,0.f,0.f};
  v8f acc[8];
  #pragma unroll
  for (int ct = 0; ct < 8; ++ct) acc[ct] = zero;

  // Prologue fill of buffer 0 (k-step 0): 2 async b128 per thread.
  ASYNC_COPY16(gfill,     sfill0);
  ASYNC_COPY16(gfill + 8, sfill0 + 8);

  #pragma unroll
  for (unsigned ki = 0; ki < 8; ++ki) {
    const unsigned kb = ki * 32;
    if (ki + 1 < 8) {             // launch fill for next k-step into other buffer
      const bf16_t* g = gfill + kb + 32;
      bf16_t* d = ((ki + 1) & 1) ? sfill1 : sfill0;
      ASYNC_COPY16(g,     d);
      ASYNC_COPY16(g + 8, d + 8);
    }
    // Async ops complete in order: <=2 outstanding => current buffer done.
    if (ki + 1 < 8) WAIT_ASYNC(2); else WAIT_ASYNC(0);
    __syncthreads();

    const unsigned abase = (rowBase + lm) * DD + kb + half * 16;
    v8bf a0 = *(const v8bf*)(Zw + abase);
    v8bf a1 = *(const v8bf*)(Zw + abase + 8);
    v16bf af = cat8(a0, a1);
    const bf16_t* sbuf = sB[ki & 1];
    #pragma unroll
    for (unsigned ct = 0; ct < 8; ++ct) {
      const bf16_t* srow = sbuf + (ct * 16 + lm) * BPAD + half * 16;
      v8bf b0 = *(const v8bf*)srow;
      v8bf b1 = *(const v8bf*)(srow + 8);
      acc[ct] = __builtin_amdgcn_wmma_f32_16x16x32_bf16(
          false, af, false, cat8(b0, b1), (short)0, acc[ct], false, false);
    }
    __syncthreads();  // all reads of this buffer done before it is refilled
  }

  // Fused epilogue
  float bet[8], qin[8];
  #pragma unroll
  for (unsigned r = 0; r < 8; ++r) {
    const unsigned m = rowBase + 8 * half + r;
    bet[r] = beta[m];
    qin[r] = 0.5f / qrsum[m];
  }
  #pragma unroll
  for (unsigned ct = 0; ct < 8; ++ct) {
    const unsigned n = colBase + ct * 16 + lm;
    #pragma unroll
    for (unsigned r = 0; r < 8; ++r) {
      const unsigned m = rowBase + 8 * half + r;
      const unsigned e = m * NN + n;            // < 64M, fits 32-bit
      float x = acc[ct][r] - bet[r];
      x = fmaxf(x, 0.f) + EPSV;
      const float p = __builtin_sqrtf(x);       // power 1/(gamma-1) = 0.5
      const float q = __builtin_nontemporal_load(Q + e);
      out[e] = p * (0.5f + qin[r] * q);
    }
  }
}

// ---------------------------------------------------------------------------
// Row-normalize out in place (deterministic: block-per-row, no atomics).
__global__ __launch_bounds__(256) void k_rownorm(float* __restrict__ out) {
  __shared__ float red[256];
  const unsigned row = blockIdx.x;
  float4* p = (float4*)out + (size_t)row * (NN / 4);
  float4 v[8];
  float s = 0.f;
  #pragma unroll
  for (int j = 0; j < 8; ++j) {
    v[j] = p[threadIdx.x + j * 256];
    s += v[j].x + v[j].y + v[j].z + v[j].w;
  }
  red[threadIdx.x] = s;
  __syncthreads();
  for (int off = 128; off > 0; off >>= 1) {
    if ((int)threadIdx.x < off) red[threadIdx.x] += red[threadIdx.x + off];
    __syncthreads();
  }
  const float inv = 1.0f / red[0];
  #pragma unroll
  for (int j = 0; j < 8; ++j) {
    v[j].x *= inv; v[j].y *= inv; v[j].z *= inv; v[j].w *= inv;
    p[threadIdx.x + j * 256] = v[j];
  }
}

// ---------------------------------------------------------------------------
extern "C" void kernel_launch(void* const* d_in, const int* in_sizes, int n_in,
                              void* d_out, int out_size, void* d_ws, size_t ws_size,
                              hipStream_t stream) {
  const float* z    = (const float*)d_in[0];  // [N,D]
  const float* W    = (const float*)d_in[1];  // [D,D]
  const float* beta = (const float*)d_in[2];  // [N,1]
  const float* Q    = (const float*)d_in[3];  // [N,N]
  float* out = (float*)d_out;

  // Workspace layout (~8.2 MiB): zb | Zw | Wt | qrsum
  char* ws = (char*)d_ws;
  bf16_t* zb    = (bf16_t*)(ws);                                  // N*D bf16
  bf16_t* Zw    = (bf16_t*)(ws + (size_t)NN * DD * 2);            // N*D bf16
  bf16_t* Wt    = (bf16_t*)(ws + (size_t)2 * NN * DD * 2);        // D*D bf16
  float*  qrsum = (float*) (ws + (size_t)2 * NN * DD * 2 + (size_t)DD * DD * 2);

  k_cvt_z  <<<(NN * DD / 4) / 256, 256, 0, stream>>>(z, zb);
  k_cvt_wt <<<(DD * DD) / 256,      256, 0, stream>>>(W, Wt);
  k_qrowsum<<<NN,                   256, 0, stream>>>(Q, qrsum);
  k_gemm_zw<<<dim3(DD / 128, NN / 128), 256, 0, stream>>>(zb, Wt, Zw);
  k_main   <<<dim3(NN / 128, NN / 128), 256, 0, stream>>>(Zw, zb, beta, Q, qrsum, out);
  k_rownorm<<<NN,                   256, 0, stream>>>(out);
}
